// SpikingDeepGalerkinNet_89249420411079
// MI455X (gfx1250) — matrix operand order repaired and checked by reading, not compiled, not run on hardware
//
#include <hip/hip_runtime.h>
#include <hip/hip_bf16.h>

typedef __attribute__((ext_vector_type(16))) _Float16 v16h;
typedef __attribute__((ext_vector_type(8)))  float    v8f;

#define T_SIM   10
#define HDIM    128
#define NLAYERS 4
#define ROWS_PER_BLOCK 128   // 8 waves * 16 rows
#define THREADS 256

// LDS layout (bytes), all 16B aligned
#define WH_OFF   0                         // f16 Wh[3][128][128] = 98304 B
#define SPK_OFF  98304                     // f16 spk[8 waves][16][128] = 32768 B
#define CUR0_OFF 106496                    // f32 cur0[8 waves][16][128] = 65536 B
#define BETA_OFF 139264                    // f32 betas[4][128] = 2048 B
#define THR_OFF  141312                    // f32 thrs[4][128]  = 2048 B
#define BH_OFF   143360                    // f32 bh[3][128]    = 1536 B
#define WOUT_OFF 144896                    // f32 wout[128]     =  512 B
#define LDS_BYTES 145408

static __device__ __forceinline__ v16h make_v16h(uint4 a, uint4 b) {
    union { uint4 u[2]; v16h v; } t;
    t.u[0] = a; t.u[1] = b;
    return t.v;
}

__global__ __launch_bounds__(THREADS, 1)
void snn_dgm_kernel(const float* __restrict__ inputs,  // (B,4)
                    const float* __restrict__ W0,      // (128,4)
                    const float* __restrict__ b0,      // (128)
                    const float* __restrict__ Wh,      // (3,128,128)
                    const float* __restrict__ bh,      // (3,128)
                    const float* __restrict__ Wout,    // (1,128)
                    const float* __restrict__ bout,    // (1)
                    const float* __restrict__ betas,   // (4,128)
                    const float* __restrict__ thrs,    // (4,128)
                    float* __restrict__ out)           // (B,1)
{
    extern __shared__ char smem[];

    const int tid  = threadIdx.x;
    const int w    = tid >> 5;        // wave id 0..7
    const int lane = tid & 31;
    const int l15  = lane & 15;
    const int hi   = lane >> 4;       // 0 for lanes 0-15, 1 for lanes 16-31
    const int rowbase = blockIdx.x * ROWS_PER_BLOCK + w * 16;

    // ---- Preload Wh (f32 -> f16) into LDS, row-major [l][n][k] ----
    {
        const float4* src = (const float4*)Wh;
        uint2* dst = (uint2*)(smem + WH_OFF);
        for (int i = tid; i < 3 * HDIM * HDIM / 4; i += THREADS) {
            float4 v = src[i];
            union { _Float16 h[4]; uint2 u; } p;
            p.h[0] = (_Float16)v.x; p.h[1] = (_Float16)v.y;
            p.h[2] = (_Float16)v.z; p.h[3] = (_Float16)v.w;
            dst[i] = p.u;
        }
        // small per-column parameter tables -> LDS (keep them OUT of VGPRs)
        float* bl = (float*)(smem + BETA_OFF);
        float* tl = (float*)(smem + THR_OFF);
        for (int i = tid; i < NLAYERS * HDIM; i += THREADS) {
            bl[i] = betas[i];
            tl[i] = thrs[i];
        }
        float* bhl = (float*)(smem + BH_OFF);
        for (int i = tid; i < (NLAYERS - 1) * HDIM; i += THREADS) bhl[i] = bh[i];
        float* wol = (float*)(smem + WOUT_OFF);
        for (int i = tid; i < HDIM; i += THREADS) wol[i] = Wout[i];
    }
    __syncthreads();

    const float* betalds = (const float*)(smem + BETA_OFF);
    const float* thrlds  = (const float*)(smem + THR_OFF);
    const float* bhlds   = (const float*)(smem + BH_OFF);
    const float* woutlds = (const float*)(smem + WOUT_OFF);
    const float bout0 = bout[0];

    // ---- cur0 = inputs @ W0^T + b0 (loop-invariant; compute once, cache in LDS) ----
    float* curlds = (float*)(smem + CUR0_OFF) + w * (16 * HDIM);
    {
        float4 w4r[8]; float b0r[8];
#pragma unroll
        for (int nt = 0; nt < 8; ++nt) {
            int col = nt * 16 + l15;
            w4r[nt] = ((const float4*)W0)[col];
            b0r[nt] = b0[col];
        }
#pragma unroll
        for (int j = 0; j < 8; ++j) {
            int rl = j + 8 * hi;
            float4 x = ((const float4*)inputs)[rowbase + rl];
#pragma unroll
            for (int nt = 0; nt < 8; ++nt) {
                float c = b0r[nt] + x.x * w4r[nt].x + x.y * w4r[nt].y
                                  + x.z * w4r[nt].z + x.w * w4r[nt].w;
                curlds[rl * HDIM + nt * 16 + l15] = c;
            }
        }
    }

    // ---- Membrane state in WMMA C-fragment layout: mem[layer][ntile] ----
    v8f mem[NLAYERS][8];
    {
        v8f z = {0.f, 0.f, 0.f, 0.f, 0.f, 0.f, 0.f, 0.f};
#pragma unroll
        for (int i = 0; i < NLAYERS; ++i)
#pragma unroll
            for (int nt = 0; nt < 8; ++nt) mem[i][nt] = z;
    }
    float oacc[8] = {0.f, 0.f, 0.f, 0.f, 0.f, 0.f, 0.f, 0.f};

    _Float16* spkbuf = (_Float16*)(smem + SPK_OFF) + w * (16 * HDIM);

    // =================== time loop ===================
#pragma unroll 1
    for (int t = 0; t < T_SIM; ++t) {
        // ---- layer 0: LIF with cached cur0; emit spikes to LDS (f16) ----
#pragma unroll
        for (int nt = 0; nt < 8; ++nt) {
            const float beta = betalds[nt * 16 + l15];
            const float thr  = thrlds[nt * 16 + l15];
#pragma unroll
            for (int j = 0; j < 8; ++j) {
                int rl = j + 8 * hi;
                float cur = curlds[rl * HDIM + nt * 16 + l15];
                float m   = mem[0][nt][j];
                float mn  = __builtin_fmaf(beta, m, cur) - ((m > thr) ? thr : 0.f);
                mem[0][nt][j] = mn;
                spkbuf[rl * HDIM + nt * 16 + l15] =
                    (mn > thr) ? (_Float16)1.0f : (_Float16)0.0f;
            }
        }
        asm volatile("s_wait_dscnt 0" ::: "memory");

        // ---- layers 1..3: spikes @ Wh^T via v_wmma_f32_16x16x32_f16 ----
#pragma unroll
        for (int li = 1; li < NLAYERS; ++li) {
            // A fragments (16x32 f16 per chunk) from spike staging buffer.
            // ISA A-layout: lanes 0-15 row M=lane, bytes [0..15]+[32..47] of
            // each 64B k-chunk; lanes 16-31 bytes [16..31]+[48..63].
            v16h afrag[4];
            {
                const char* abase = (const char*)spkbuf + l15 * (HDIM * 2)
                                    + ((lane < 16) ? 0 : 16);
#pragma unroll
                for (int c = 0; c < 4; ++c) {
                    uint4 a0 = *(const uint4*)(abase + c * 64);
                    uint4 a1 = *(const uint4*)(abase + c * 64 + 32);
                    afrag[c] = make_v16h(a0, a1);
                }
            }
            const char* wlayer = smem + WH_OFF + (li - 1) * (HDIM * HDIM * 2);

#pragma unroll
            for (int nt = 0; nt < 8; ++nt) {
                // Accumulator chain starts from the inline C=0 form (no VGPR
                // splat, no tied-C copy); bias is folded into the LIF epilogue.
                v8f acc = {0.f, 0.f, 0.f, 0.f, 0.f, 0.f, 0.f, 0.f};
                // B column n = Wh row n (cur = spk @ Wh^T); ISA B-layout:
                // lanes 0-15 hold K 0..7 + 16..23, lanes 16-31
                // hold K 8..15 + 24..31 (same as A).
                const char* bbase = wlayer + (nt * 16 + l15) * (HDIM * 2)
                                    + ((lane < 16) ? 0 : 16);
#pragma unroll
                for (int c = 0; c < 4; ++c) {
                    uint4 q0 = *(const uint4*)(bbase + c * 64);
                    uint4 q1 = *(const uint4*)(bbase + c * 64 + 32);
                    v16h bf = make_v16h(q0, q1);
                    acc = __builtin_amdgcn_wmma_f32_16x16x32_f16(
                        false, afrag[c], false, bf, (short)0, acc, false, false);
                }
                // LIF update for this N-tile, in C-fragment layout.
                // mn = beta*m + (acc + bias) - reset
                //    = fma(beta, m, acc) + select(m>thr, bias-thr, bias)
                const float beta  = betalds[li * HDIM + nt * 16 + l15];
                const float thr   = thrlds[li * HDIM + nt * 16 + l15];
                const float bias  = bhlds[(li - 1) * HDIM + nt * 16 + l15];
                const float biasr = bias - thr;   // bias minus reset amount
#pragma unroll
                for (int j = 0; j < 8; ++j) {
                    float m  = mem[li][nt][j];
                    float mn = __builtin_fmaf(beta, m, acc[j])
                             + ((m > thr) ? biasr : bias);
                    mem[li][nt][j] = mn;
                    if (li < 3)  // layer-3 spikes are never consumed
                        spkbuf[(j + 8 * hi) * HDIM + nt * 16 + l15] =
                            (mn > thr) ? (_Float16)1.0f : (_Float16)0.0f;
                }
            }
            if (li < 3) asm volatile("s_wait_dscnt 0" ::: "memory");
        }

        // ---- output accumulation: oacc += mem3 . Wout ----
#pragma unroll
        for (int nt = 0; nt < 8; ++nt) {
            const float wv = woutlds[nt * 16 + l15];
#pragma unroll
            for (int j = 0; j < 8; ++j)
                oacc[j] = __builtin_fmaf(mem[3][nt][j], wv, oacc[j]);
        }
    }

    // ---- reduce partial dots across the 16 lanes sharing a row, write out ----
#pragma unroll
    for (int j = 0; j < 8; ++j) {
        float v = oacc[j];
        v += __shfl_xor(v, 1, 32);
        v += __shfl_xor(v, 2, 32);
        v += __shfl_xor(v, 4, 32);
        v += __shfl_xor(v, 8, 32);
        if (l15 == 0)  // lane 0 -> row j, lane 16 -> row j+8
            out[rowbase + j + 8 * hi] = v * (1.0f / (float)T_SIM) + bout0;
    }
}

extern "C" void kernel_launch(void* const* d_in, const int* in_sizes, int n_in,
                              void* d_out, int out_size, void* d_ws, size_t ws_size,
                              hipStream_t stream) {
    const float* inputs = (const float*)d_in[0];
    const float* W0     = (const float*)d_in[1];
    const float* b0     = (const float*)d_in[2];
    const float* Wh     = (const float*)d_in[3];
    const float* bh     = (const float*)d_in[4];
    const float* Wout   = (const float*)d_in[5];
    const float* bout   = (const float*)d_in[6];
    const float* betas  = (const float*)d_in[7];
    const float* thrs   = (const float*)d_in[8];
    float* out = (float*)d_out;

    const int B = in_sizes[0] / 4;                    // (B, 1+d) with d=3
    const int nblocks = (B + ROWS_PER_BLOCK - 1) / ROWS_PER_BLOCK;

    (void)hipFuncSetAttribute((const void*)snn_dgm_kernel,
                              hipFuncAttributeMaxDynamicSharedMemorySize,
                              (int)LDS_BYTES);

    hipLaunchKernelGGL(snn_dgm_kernel, dim3(nblocks), dim3(THREADS), LDS_BYTES,
                       stream, inputs, W0, b0, Wh, bh, Wout, bout, betas, thrs,
                       out);
}
